// GAT_15187004358790
// MI455X (gfx1250) — matrix operand order repaired and checked by reading, not compiled, not run on hardware
//
#include <hip/hip_runtime.h>
#include <hip/hip_bf16.h>
#include <stdint.h>

#define NNODES 100000
#define NEDGES 1600000
#define NGRAPH 64
#define EPRIME (NEDGES + NNODES)

typedef __attribute__((ext_vector_type(16))) _Float16 v16h;
typedef __attribute__((ext_vector_type(8)))  _Float16 v8h;
typedef __attribute__((ext_vector_type(8)))  float    v8f;
typedef __attribute__((ext_vector_type(2)))  float    v2f;

#if defined(__has_builtin)
#if __has_builtin(__builtin_amdgcn_wmma_f32_16x16x4_f32)
#define HAVE_WMMA_F32K4 1
#else
#define HAVE_WMMA_F32K4 0
#endif
#else
#define HAVE_WMMA_F32K4 0
#endif

// ---- order-preserving float -> uint key for atomic max (matches segment_max with -inf init)
static __device__ __forceinline__ unsigned fkey(float f) {
  unsigned u = __float_as_uint(f);
  return (u & 0x80000000u) ? ~u : (u | 0x80000000u);
}
static __device__ __forceinline__ float fdecode(unsigned k) {
  unsigned u = (k & 0x80000000u) ? (k & 0x7FFFFFFFu) : ~k;
  return __uint_as_float(u);
}
#define MKEY_NEG_INF 0x007FFFFFu   // fkey(-inf)

// ======================= small utility kernels =======================
__global__ void cast_edges_k(const long long* __restrict__ ei,
                             int* __restrict__ s32, int* __restrict__ d32) {
  int e = blockIdx.x * blockDim.x + threadIdx.x;
  if (e < NEDGES) {
    s32[e] = (int)ei[e];
    d32[e] = (int)ei[(size_t)NEDGES + e];
  }
}

__global__ void fill_u32_k(unsigned* __restrict__ p, unsigned v, int n) {
  int i = blockIdx.x * blockDim.x + threadIdx.x;
  if (i < n) p[i] = v;
}

__global__ void cvt_f16_k(const float* __restrict__ s, _Float16* __restrict__ d, int n) {
  int i = blockIdx.x * blockDim.x + threadIdx.x;
  if (i < n) d[i] = (_Float16)s[i];
}

// pad W4 [64,4] -> [64,16] f16, zeros beyond col 3
__global__ void cvt_w4_k(const float* __restrict__ W4, _Float16* __restrict__ d) {
  int i = blockIdx.x * blockDim.x + threadIdx.x;
  if (i < 64 * 16) {
    int k = i >> 4, n = i & 15;
    d[i] = (n < 4) ? (_Float16)W4[k * 4 + n] : (_Float16)0.0f;
  }
}

// ======================= layer 1 transform: [N,4] @ [4,64] =======================
#if HAVE_WMMA_F32K4
// V_WMMA_F32_16X16X4_F32: A 16x4 f32 (lane l: row l&15, K = 2*(l>>4)+{0,1}),
// B 4x16 f32 (lane l: col l&15, same K pattern). 4 waves/block = 4 col tiles.
__global__ void gemm1_wmma_k(const float* __restrict__ x, const float* __restrict__ W1,
                             float* __restrict__ out, int mtPerBlock) {
  const int lane = threadIdx.x & 31;
  const int wave = threadIdx.x >> 5;   // col tile
  const int col = lane & 15;
  const int hi = lane >> 4;
  const int n0 = wave * 16;

  v2f b;
  b.x = W1[(2 * hi) * 64 + n0 + col];
  b.y = W1[(2 * hi + 1) * 64 + n0 + col];

  int mt0 = blockIdx.x * mtPerBlock;
  int tiles = (NNODES / 16) - mt0;
  if (tiles <= 0) return;
  if (tiles > mtPerBlock) tiles = mtPerBlock;

  const float* ap = x + (size_t)(mt0 * 16 + col) * 4 + 2 * hi;
  float* cp = out + (size_t)(mt0 * 16 + 8 * hi) * 64 + n0 + col;
  for (int t = 0; t < tiles; ++t) {
    v2f a = *(const v2f*)ap;
    v8f acc = {};
    acc = __builtin_amdgcn_wmma_f32_16x16x4_f32(false, a, false, b, (short)0, acc, false, false);
#pragma unroll
    for (int r = 0; r < 8; ++r) cp[r * 64] = acc[r];
    ap += 16 * 4;
    cp += 16 * 64;
  }
}
#endif

// VALU fallback (also the bandwidth-roofline baseline: reads 1.6MB, writes 25.6MB)
__global__ void gemm1_valu_k(const float* __restrict__ x, const float* __restrict__ W1,
                             float* __restrict__ out) {
  int t = blockIdx.x * blockDim.x + threadIdx.x;
  if (t >= NNODES * 64) return;
  int n = t >> 6, c = t & 63;
  float4 xv = *(const float4*)(x + (size_t)n * 4);
  out[t] = xv.x * W1[c] + xv.y * W1[64 + c] + xv.z * W1[128 + c] + xv.w * W1[192 + c];
}

// ======================= WMMA GEMM: A[nRows,64]f16 @ B[64,16*NWAVES]f16 -> C[nRows,LDC]f32 ====
// Compile-time LDC -> C tile store is one address + 8 stores at immediate offsets (r*LDC*4 B).
// B fragments hoisted out of the M loop; 2 WMMAs (K=32 each) per 16-row tile.
template <int NWAVES, int LDC, int NSTORE>
__global__ void gemm_f16_k64_t(const _Float16* __restrict__ A, const _Float16* __restrict__ B,
                               float* __restrict__ C, int nRows, int mtPerBlock) {
  const int lane = threadIdx.x & 31;
  const int wave = threadIdx.x >> 5;
  const int ldb = NWAVES * 16;
  const int n0 = wave * 16;
  const int col = lane & 15;
  const int hi = lane >> 4;

  // B (32x16 f16): lane l holds col l&15, K = 16*(l>>4) + i, i = 0..15
  v16h b0, b1;
  {
    const int kb = 16 * hi;
#pragma unroll
    for (int i = 0; i < 16; ++i) {
      b0[i] = B[(kb + i) * ldb + n0 + col];
      b1[i] = B[(32 + kb + i) * ldb + n0 + col];
    }
  }

  int mt0 = blockIdx.x * mtPerBlock;
  int tiles = (nRows / 16) - mt0;
  if (tiles <= 0) return;
  if (tiles > mtPerBlock) tiles = mtPerBlock;

  // A (16x32 f16): lane l holds row l&15, K = 16*(l>>4) + i, i = 0..15
  const int kqa = 16 * hi;
  const _Float16* ap = A + (size_t)(mt0 * 16 + col) * 64 + kqa;
  float* cp = C + (size_t)(mt0 * 16 + 8 * hi) * LDC + n0 + col;
  const bool doStore = (NSTORE >= NWAVES * 16) || (n0 + col < NSTORE);

  for (int t = 0; t < tiles; ++t) {
    v8h a00 = *(const v8h*)(ap);
    v8h a01 = *(const v8h*)(ap + 8);
    v8h a10 = *(const v8h*)(ap + 32);
    v8h a11 = *(const v8h*)(ap + 40);
    v16h a0, a1;
#pragma unroll
    for (int i = 0; i < 8; ++i) {
      a0[i] = a00[i]; a0[8 + i] = a01[i];
      a1[i] = a10[i]; a1[8 + i] = a11[i];
    }
    v8f acc = {};
    acc = __builtin_amdgcn_wmma_f32_16x16x32_f16(false, a0, false, b0, (short)0, acc, false, false);
    acc = __builtin_amdgcn_wmma_f32_16x16x32_f16(false, a1, false, b1, (short)0, acc, false, false);
    if (doStore) {
#pragma unroll
      for (int r = 0; r < 8; ++r) cp[r * LDC] = acc[r];
    }
    ap += 16 * 64;
    cp += 16 * LDC;
  }
}

// ======================= attention scores s_src/s_dst per (node, head) =======================
__global__ void scores_k(const float* __restrict__ hT, const float* __restrict__ aS,
                         const float* __restrict__ aD, float* __restrict__ sS,
                         float* __restrict__ sD, int H, int C) {
  int t = blockIdx.x * blockDim.x + threadIdx.x;
  if (t >= NNODES * H) return;
  int n = t / H, h = t % H;
  const float* hp = hT + (size_t)n * (H * C) + h * C;
  const float* as = aS + h * C;
  const float* ad = aD + h * C;
  float s0 = 0.f, s1 = 0.f;
  for (int c = 0; c < C; c += 4) {
    float4 hv = *(const float4*)(hp + c);
    float4 av = *(const float4*)(as + c);
    float4 dv = *(const float4*)(ad + c);
    s0 += hv.x * av.x + hv.y * av.y + hv.z * av.z + hv.w * av.w;
    s1 += hv.x * dv.x + hv.y * dv.y + hv.z * dv.z + hv.w * dv.w;
  }
  sS[t] = s0;
  sD[t] = s1;
}

// ======================= edge passes (self loops = edge ids >= NEDGES) =======================
__global__ void edge_max_k(const int* __restrict__ s32, const int* __restrict__ d32,
                           const float* __restrict__ sS, const float* __restrict__ sD,
                           unsigned* __restrict__ mkey, int H) {
  long t = (long)blockIdx.x * blockDim.x + threadIdx.x;
  if (t >= (long)EPRIME * H) return;
  int h = (int)(t % H);
  long e = t / H;
  int s = (e < NEDGES) ? s32[e] : (int)(e - NEDGES);
  int d = (e < NEDGES) ? d32[e] : s;
  float v = sS[s * H + h] + sD[d * H + h];
  v = (v > 0.f) ? v : 0.2f * v;                 // leaky_relu(0.2)
  atomicMax(&mkey[d * H + h], fkey(v));
}

__global__ void edge_denom_k(const int* __restrict__ s32, const int* __restrict__ d32,
                             const float* __restrict__ sS, const float* __restrict__ sD,
                             const unsigned* __restrict__ mkey, float* __restrict__ denom, int H) {
  long t = (long)blockIdx.x * blockDim.x + threadIdx.x;
  if (t >= (long)EPRIME * H) return;
  int h = (int)(t % H);
  long e = t / H;
  int s = (e < NEDGES) ? s32[e] : (int)(e - NEDGES);
  int d = (e < NEDGES) ? d32[e] : s;
  float v = sS[s * H + h] + sD[d * H + h];
  v = (v > 0.f) ? v : 0.2f * v;
  float m = fdecode(mkey[d * H + h]);
  if (m < -3.0e38f) m = 0.f;                    // isfinite guard as in reference
  atomicAdd(&denom[d * H + h], __expf(v - m));
}

__global__ void edge_agg_k(const int* __restrict__ s32, const int* __restrict__ d32,
                           const float* __restrict__ sS, const float* __restrict__ sD,
                           const unsigned* __restrict__ mkey, const float* __restrict__ denom,
                           const float* __restrict__ hT, float* __restrict__ out,
                           int H, int C, int chunk) {
  long t = (long)blockIdx.x * blockDim.x + threadIdx.x;
  int nCh = C / chunk;
  if (t >= (long)EPRIME * H * nCh) return;
  int ci = (int)(t % nCh);
  long eh = t / nCh;
  int h = (int)(eh % H);
  long e = eh / H;
  int s = (e < NEDGES) ? s32[e] : (int)(e - NEDGES);
  int d = (e < NEDGES) ? d32[e] : s;
  int F = H * C;
  const float* hp = hT + (size_t)s * F + h * C + ci * chunk;
  __builtin_prefetch(hp, 0, 0);                 // global_prefetch_b8: overlap gather with alpha calc
  float v = sS[s * H + h] + sD[d * H + h];
  v = (v > 0.f) ? v : 0.2f * v;
  float m = fdecode(mkey[d * H + h]);
  if (m < -3.0e38f) m = 0.f;
  float alpha = __expf(v - m) / (denom[d * H + h] + 1e-16f);
  float* op = out + (size_t)d * F + h * C + ci * chunk;
  for (int c = 0; c < chunk; c += 4) {
    float4 hv = *(const float4*)(hp + c);
    atomicAdd(op + c + 0, alpha * hv.x);
    atomicAdd(op + c + 1, alpha * hv.y);
    atomicAdd(op + c + 2, alpha * hv.z);
    atomicAdd(op + c + 3, alpha * hv.w);
  }
}

// ======================= epilogue: +bias, optional ELU, optional f16 copy =======================
__global__ void epilogue_k(float* __restrict__ buf, const float* __restrict__ bias, int F,
                           int doElu, _Float16* __restrict__ h16) {
  int t = blockIdx.x * blockDim.x + threadIdx.x;
  if (t >= NNODES * F) return;
  float v = buf[t] + bias[t % F];
  if (doElu) v = (v > 0.f) ? v : expm1f(v);
  buf[t] = v;
  if (h16) h16[t] = (_Float16)v;
}

// ======================= pooling + head =======================
__global__ void pool_k(const long long* __restrict__ batch, const float* __restrict__ h,
                       float* __restrict__ sums, float* __restrict__ cnt) {
  int n = blockIdx.x * blockDim.x + threadIdx.x;
  if (n >= NNODES) return;
  int g = (int)batch[n];
  atomicAdd(&cnt[g], 1.0f);
#pragma unroll
  for (int c = 0; c < 4; ++c) atomicAdd(&sums[g * 4 + c], h[(size_t)n * 4 + c]);
}

__global__ void head_k(const float* __restrict__ sums, const float* __restrict__ cnt,
                       const float* __restrict__ Wl, const float* __restrict__ bl,
                       float* __restrict__ out) {
  int g = blockIdx.x * blockDim.x + threadIdx.x;
  if (g >= NGRAPH) return;
  float cc = fmaxf(cnt[g], 1.0f);
  float acc = bl[0];
#pragma unroll
  for (int c = 0; c < 4; ++c) acc += (sums[g * 4 + c] / cc) * Wl[c];
  out[g] = acc;
}

// ======================= host orchestration =======================
static inline dim3 grid1(long n, int b) { return dim3((unsigned)((n + b - 1) / b)); }

extern "C" void kernel_launch(void* const* d_in, const int* in_sizes, int n_in,
                              void* d_out, int out_size, void* d_ws, size_t ws_size,
                              hipStream_t stream) {
  (void)in_sizes; (void)n_in; (void)out_size; (void)ws_size;
  const float*     x      = (const float*)d_in[0];
  const long long* ei     = (const long long*)d_in[1];
  const long long* batch  = (const long long*)d_in[2];
  const float* W1    = (const float*)d_in[3];
  const float* aS1   = (const float*)d_in[4];
  const float* aD1   = (const float*)d_in[5];
  const float* b1    = (const float*)d_in[6];
  const float* W2    = (const float*)d_in[7];
  const float* aS2   = (const float*)d_in[8];
  const float* aD2   = (const float*)d_in[9];
  const float* b2    = (const float*)d_in[10];
  const float* W4    = (const float*)d_in[11];
  const float* aS4   = (const float*)d_in[12];
  const float* aD4   = (const float*)d_in[13];
  const float* b4    = (const float*)d_in[14];
  const float* Wl    = (const float*)d_in[15];
  const float* bl    = (const float*)d_in[16];
  float* out = (float*)d_out;

  // ---- carve workspace (~90 MB)
  char* ws = (char*)d_ws;
  auto carve = [&](size_t bytes) -> char* {
    char* p = ws;
    ws += (bytes + 255) & ~(size_t)255;
    return p;
  };
  int*       s32   = (int*)carve((size_t)NEDGES * 4);
  int*       d32   = (int*)carve((size_t)NEDGES * 4);
  float*     buf0  = (float*)carve((size_t)NNODES * 64 * 4);  // transformed features h = X@W
  float*     buf1  = (float*)carve((size_t)NNODES * 64 * 4);  // aggregation target / next input
  _Float16*  hH    = (_Float16*)carve((size_t)NNODES * 64 * 2);
  float*     sS    = (float*)carve((size_t)NNODES * 8 * 4);
  float*     sD    = (float*)carve((size_t)NNODES * 8 * 4);
  unsigned*  mkey  = (unsigned*)carve((size_t)NNODES * 8 * 4);
  float*     denom = (float*)carve((size_t)NNODES * 8 * 4);
  _Float16*  W2h   = (_Float16*)carve(64 * 64 * 2);
  _Float16*  W4h   = (_Float16*)carve(64 * 16 * 2);
  float*     psum  = (float*)carve(NGRAPH * 4 * 4);
  float*     pcnt  = (float*)carve(NGRAPH * 4);

  const int mTiles = NNODES / 16;          // 6250 exact
  const int MT = 4;                        // row tiles per block in WMMA gemms
  const int gemmBlocks = (mTiles + MT - 1) / MT;

  // edge-softmax + aggregate wrapper: hT (scores source + message payload) -> outBuf
  auto gat_edges = [&](const float* hT, float* outBuf, const float* aSrc, const float* aDst,
                       int H, int C, int chunk) {
    scores_k<<<grid1((long)NNODES * H, 256), 256, 0, stream>>>(hT, aSrc, aDst, sS, sD, H, C);
    fill_u32_k<<<grid1((long)NNODES * H, 256), 256, 0, stream>>>(mkey, MKEY_NEG_INF, NNODES * H);
    hipMemsetAsync(denom, 0, (size_t)NNODES * H * 4, stream);
    hipMemsetAsync(outBuf, 0, (size_t)NNODES * H * C * 4, stream);
    long nEH = (long)EPRIME * H;
    edge_max_k<<<grid1(nEH, 256), 256, 0, stream>>>(s32, d32, sS, sD, mkey, H);
    edge_denom_k<<<grid1(nEH, 256), 256, 0, stream>>>(s32, d32, sS, sD, mkey, denom, H);
    long nAgg = nEH * (C / chunk);
    edge_agg_k<<<grid1(nAgg, 256), 256, 0, stream>>>(s32, d32, sS, sD, mkey, denom, hT, outBuf,
                                                     H, C, chunk);
  };

  // ---- prep
  cast_edges_k<<<grid1(NEDGES, 256), 256, 0, stream>>>(ei, s32, d32);
  cvt_f16_k<<<grid1(64 * 64, 256), 256, 0, stream>>>(W2, W2h, 64 * 64);
  cvt_w4_k<<<grid1(64 * 16, 256), 256, 0, stream>>>(W4, W4h);

  // ---- layer 1: GAT(4 -> 8 heads x 8), elu
#if HAVE_WMMA_F32K4
  gemm1_wmma_k<<<dim3(gemmBlocks), 128, 0, stream>>>(x, W1, buf0, MT);
#else
  gemm1_valu_k<<<grid1((long)NNODES * 64, 256), 256, 0, stream>>>(x, W1, buf0);
#endif
  gat_edges(buf0, buf1, aS1, aD1, /*H=*/8, /*C=*/8, /*chunk=*/8);
  epilogue_k<<<grid1((long)NNODES * 64, 256), 256, 0, stream>>>(buf1, b1, 64, /*elu=*/1, hH);

  // ---- layers 2 & 3: GAT(64 -> 64), same weights, no activation between
  for (int it = 0; it < 2; ++it) {
    gemm_f16_k64_t<4, 64, 64><<<dim3(gemmBlocks), 128, 0, stream>>>(hH, W2h, buf0, NNODES, MT);
    gat_edges(buf0, buf1, aS2, aD2, /*H=*/1, /*C=*/64, /*chunk=*/16);
    epilogue_k<<<grid1((long)NNODES * 64, 256), 256, 0, stream>>>(buf1, b2, 64, /*elu=*/0, hH);
  }

  // ---- layer 4: GAT(64 -> 4), heads=1 (mean over 1 head == identity)
  gemm_f16_k64_t<1, 4, 4><<<dim3(gemmBlocks), 32, 0, stream>>>(hH, W4h, buf0, NNODES, MT);
  gat_edges(buf0, buf1, aS4, aD4, /*H=*/1, /*C=*/4, /*chunk=*/4);
  epilogue_k<<<grid1((long)NNODES * 4, 256), 256, 0, stream>>>(buf1, b4, 4, /*elu=*/0, (_Float16*)0);

  // ---- global mean pool + linear head
  hipMemsetAsync(psum, 0, NGRAPH * 4 * 4, stream);
  hipMemsetAsync(pcnt, 0, NGRAPH * 4, stream);
  pool_k<<<grid1(NNODES, 256), 256, 0, stream>>>(batch, buf1, psum, pcnt);
  head_k<<<grid1(NGRAPH, 64), 64, 0, stream>>>(psum, pcnt, Wl, bl, out);
}
